// CausalSelfAttention_48189533061120
// MI455X (gfx1250) — compile-verified, matrix operand order checked by reading, not run on hardware
//
#include <hip/hip_runtime.h>
#include <hip/hip_bf16.h>
#include <stdint.h>

typedef __attribute__((ext_vector_type(16))) __bf16 v16bf;
typedef __attribute__((ext_vector_type(8)))  float  v8f;

union FragU { v16bf v; uint4 u4[2]; unsigned short s[16]; };
union U4S  { uint4 u; unsigned short s[8]; };

__device__ __forceinline__ unsigned short f2bf(float f) {
  union { float f; uint32_t u; } c; c.f = f;
  uint32_t r = c.u + 0x7FFFu + ((c.u >> 16) & 1u);
  return (unsigned short)(r >> 16);
}

// Async 16-byte global->LDS copy (per-lane gather/scatter), ASYNCcnt-tracked.
__device__ __forceinline__ void async_cp16(uint32_t lds_off, uint64_t gaddr) {
  asm volatile("global_load_async_to_lds_b128 %0, %1, off"
               :: "v"(lds_off), "v"(gaddr) : "memory");
}
__device__ __forceinline__ void wait_async0() {
  asm volatile("s_wait_asynccnt 0" ::: "memory");
}

// ---------------- fp32 -> bf16 conversion ----------------
__global__ __launch_bounds__(256) void cvt_f32_bf16(const float* __restrict__ in,
                                                    unsigned short* __restrict__ out, int n) {
  int i = blockIdx.x * blockDim.x + threadIdx.x;
  int st = gridDim.x * blockDim.x;
  for (; i < n; i += st) out[i] = f2bf(in[i]);
}

// ---------------- bf16 GEMM: D = A[MxK] * B[KxN] + bias ----------------
// Double-buffered LDS; A tile staged via async-to-LDS, B tile via scatter stores.
// MODE 0: scatter q/k/v bf16 epilogue (N=3072). MODE 1: f32 store to out.
template <int MODE>
__global__ __launch_bounds__(256) void gemm_bf16(
    const unsigned short* __restrict__ A, const unsigned short* __restrict__ Bm,
    const float* __restrict__ bias, int M, int N, int K,
    unsigned short* __restrict__ q, unsigned short* __restrict__ k,
    unsigned short* __restrict__ v, float* __restrict__ outp) {
  __shared__ __align__(16) unsigned short Alds[2][8 * 512];  // fragment-ordered
  __shared__ __align__(16) unsigned short Blds[2][8 * 512];

  const int tid = threadIdx.x;
  const int wid = tid >> 5, lane = tid & 31;
  const int lo = lane & 15, hi = lane >> 4;
  const int mBase = blockIdx.y * 128;
  const int nBase = blockIdx.x * 128;
  const int wm = (wid & 3) * 32;   // wave sub-tile: 32 rows
  const int wn = (wid >> 2) * 64;  // x 64 cols

  v8f acc[2][4];
#pragma unroll
  for (int i = 0; i < 2; ++i)
#pragma unroll
    for (int j = 0; j < 4; ++j) acc[i][j] = (v8f){0.f,0.f,0.f,0.f,0.f,0.f,0.f,0.f};

  const int amm = tid & 127;            // A tile row
  const int akc = (tid >> 7) * 16;      // A k-chunk base (0 or 16)
  const int bkk = tid & 31;             // B tile row (k)
  const int bnc = (tid >> 5) * 16;      // B n-chunk base

  // stage A tile (128x32) into fragment-ordered LDS via async copies
  auto stageA = [&](int buf, int kbase) {
    const uint64_t g = (uint64_t)(uintptr_t)(A + (size_t)(mBase + amm) * K + kbase + akc);
    const int s = amm >> 4, mm = amm & 15;
    {  // chunk k0 = akc  -> lane lo half, e0 = (akc&16)?8:0
      const int e0 = (akc & 16) ? 8 : 0;
      async_cp16((uint32_t)(uintptr_t)&Alds[buf][s * 512 + mm * 16 + e0], g);
    }
    {  // chunk k0 = akc+8 -> lane hi half
      const int k0 = akc + 8;
      const int e0 = (k0 & 16) ? 8 : 0;
      async_cp16((uint32_t)(uintptr_t)&Alds[buf][s * 512 + (mm + 16) * 16 + e0], g + 16);
    }
  };
  // stage B tile (32x128) into fragment-ordered LDS (element scatter)
  auto stageB = [&](int buf, int kbase) {
    const uint4* gB =
        reinterpret_cast<const uint4*>(Bm + (size_t)(kbase + bkk) * N + nBase + bnc);
    U4S b0, b1; b0.u = gB[0]; b1.u = gB[1];
    const int s = bnc >> 4;
    const int e = (bkk & 16) ? ((bkk & 7) + 8) : (bkk & 7);
    const int lg = 16 * ((bkk >> 3) & 1);
#pragma unroll
    for (int i = 0; i < 8; ++i) Blds[buf][s * 512 + (i + lg) * 16 + e] = b0.s[i];
#pragma unroll
    for (int i = 0; i < 8; ++i) Blds[buf][s * 512 + (8 + i + lg) * 16 + e] = b1.s[i];
  };

  stageA(0, 0);
  stageB(0, 0);
  wait_async0();
  __syncthreads();

  int cur = 0;
  for (int kb = 0; kb < K; kb += 32) {
    const int nxt = cur ^ 1;
    if (kb + 32 < K) {  // prefetch next tile into alternate buffer
      stageA(nxt, kb + 32);
      stageB(nxt, kb + 32);
      __builtin_prefetch(A + (size_t)(mBase + amm) * K + kb + 64 + akc, 0, 1);
      __builtin_prefetch(Bm + (size_t)(kb + 64 + bkk) * N + nBase + bnc, 0, 1);
    }

    // ---- fragments + WMMA from current buffer ----
    FragU af[2], bf[4];
    const int s0 = wm >> 4, n0 = wn >> 4;
#pragma unroll
    for (int i = 0; i < 2; ++i) {
      const uint4* p = reinterpret_cast<const uint4*>(&Alds[cur][(s0 + i) * 512 + lane * 16]);
      af[i].u4[0] = p[0]; af[i].u4[1] = p[1];
    }
#pragma unroll
    for (int j = 0; j < 4; ++j) {
      const uint4* p = reinterpret_cast<const uint4*>(&Blds[cur][(n0 + j) * 512 + lane * 16]);
      bf[j].u4[0] = p[0]; bf[j].u4[1] = p[1];
    }
#pragma unroll
    for (int i = 0; i < 2; ++i)
#pragma unroll
      for (int j = 0; j < 4; ++j)
        acc[i][j] = __builtin_amdgcn_wmma_f32_16x16x32_bf16(
            false, af[i].v, false, bf[j].v, (short)0, acc[i][j], false, false);

    wait_async0();      // async stores to nxt buffer complete
    __syncthreads();    // all waves done reading cur, done writing nxt
    cur = nxt;
  }

  // ---- epilogue ----
#pragma unroll
  for (int i = 0; i < 2; ++i)
#pragma unroll
    for (int j = 0; j < 4; ++j)
#pragma unroll
      for (int g = 0; g < 8; ++g) {
        int row = mBase + wm + i * 16 + g + 8 * hi;
        int col = nBase + wn + j * 16 + lo;
        float val = acc[i][j][g] + bias[col];
        if (MODE == 0) {
          int b = row >> 11, t = row & 2047;
          int which = col >> 10, cc = col & 1023;
          int h = cc >> 6, dd = cc & 63;
          size_t idx = (((size_t)b * 16 + h) * 2048 + t) * 64 + dd;
          unsigned short bv = f2bf(val);
          if (which == 0) q[idx] = bv;
          else if (which == 1) k[idx] = bv;
          else v[idx] = bv;
        } else {
          outp[(size_t)row * N + col] = val;
        }
      }
}

// ---------------- flash attention (bf16 WMMA, f32 softmax) ----------------
// Q,K,V: [B*H, T, 64] bf16.  Out: [B*T, 1024] bf16 (row = b*T+t, col = h*64+d).
__global__ __launch_bounds__(256) void flash_attn(
    const unsigned short* __restrict__ Q, const unsigned short* __restrict__ K,
    const unsigned short* __restrict__ V, unsigned short* __restrict__ Ob, int T) {
  __shared__ __align__(16) unsigned short Klds[64 * 80];      // [key][d], stride 80
  __shared__ __align__(16) unsigned short Vt[64 * 80];        // [d][key], stride 80
  __shared__ __align__(16) unsigned short Plds[8 * 16 * 64];  // per-wave P, stride 64

  const int tid = threadIdx.x, wid = tid >> 5, lane = tid & 31;
  const int lo = lane & 15, hi = lane >> 4;
  const int bh = blockIdx.y;                 // b*16 + h
  const int b = bh >> 4, h = bh & 15;
  const int qTile = blockIdx.x * 128;
  const int r0 = qTile + wid * 16;           // wave's 16 query rows
  const size_t headBase = (size_t)bh * T * 64;
  const float scale = 0.125f;                // 1/sqrt(64)

  // Q fragments (A-layout), held for the whole kv loop
  FragU qf[2];
  {
    const size_t rb = headBase + (size_t)(r0 + lo) * 64;
    qf[0].u4[0] = *(const uint4*)&Q[rb + 0  + 8 * hi];
    qf[0].u4[1] = *(const uint4*)&Q[rb + 16 + 8 * hi];
    qf[1].u4[0] = *(const uint4*)&Q[rb + 32 + 8 * hi];
    qf[1].u4[1] = *(const uint4*)&Q[rb + 48 + 8 * hi];
  }

  v8f o[4];
#pragma unroll
  for (int j = 0; j < 4; ++j) o[j] = (v8f){0.f,0.f,0.f,0.f,0.f,0.f,0.f,0.f};
  float mrow[8], lrow[8];
#pragma unroll
  for (int g = 0; g < 8; ++g) { mrow[g] = -1e30f; lrow[g] = 0.f; }

  const int jbMax = (qTile + 127) >> 6;
  for (int jb = 0; jb <= jbMax; ++jb) {
    const int kvBase = jb * 64;
    // ---- stage K [key][d] (async direct copy) and V transposed [d][key] ----
    {
      const int kr = tid & 63, dc = (tid >> 6) * 16;
      const uint64_t gk =
          (uint64_t)(uintptr_t)(K + headBase + (size_t)(kvBase + kr) * 64 + dc);
      async_cp16((uint32_t)(uintptr_t)&Klds[kr * 80 + dc], gk);
      async_cp16((uint32_t)(uintptr_t)&Klds[kr * 80 + dc + 8], gk + 16);

      const uint4* gv = (const uint4*)&V[headBase + (size_t)(kvBase + kr) * 64 + dc];
      U4S v0, v1; v0.u = gv[0]; v1.u = gv[1];
#pragma unroll
      for (int i = 0; i < 8; ++i) Vt[(dc + i) * 80 + kr] = v0.s[i];
#pragma unroll
      for (int i = 0; i < 8; ++i) Vt[(dc + 8 + i) * 80 + kr] = v1.s[i];
    }
    wait_async0();
    __syncthreads();

    // ---- S = Q K^T (16 x 64 per wave) ----
    v8f s[4];
#pragma unroll
    for (int sub = 0; sub < 4; ++sub) {
      s[sub] = (v8f){0.f,0.f,0.f,0.f,0.f,0.f,0.f,0.f};
      const int n = sub * 16 + lo;
#pragma unroll
      for (int kc = 0; kc < 2; ++kc) {
        FragU bfr;
        bfr.u4[0] = *(const uint4*)&Klds[n * 80 + kc * 32 + 8 * hi];
        bfr.u4[1] = *(const uint4*)&Klds[n * 80 + kc * 32 + 16 + 8 * hi];
        s[sub] = __builtin_amdgcn_wmma_f32_16x16x32_bf16(
            false, qf[kc].v, false, bfr.v, (short)0, s[sub], false, false);
      }
    }

    // ---- causal mask + online softmax (row lives in one 16-lane half) ----
#pragma unroll
    for (int g = 0; g < 8; ++g) {
      const int qi = r0 + g + 8 * hi;
      float mx = -1e30f;
#pragma unroll
      for (int sub = 0; sub < 4; ++sub) {
        const int ki = kvBase + sub * 16 + lo;
        float val = s[sub][g] * scale;
        if (ki > qi) val = -1e30f;
        s[sub][g] = val;
        mx = fmaxf(mx, val);
      }
#pragma unroll
      for (int off = 1; off < 16; off <<= 1) mx = fmaxf(mx, __shfl_xor(mx, off, 16));
      const float mnew = fmaxf(mrow[g], mx);
      const float alpha = __expf(mrow[g] - mnew);
      float rs = 0.f;
#pragma unroll
      for (int sub = 0; sub < 4; ++sub) {
        const float p = __expf(s[sub][g] - mnew);
        s[sub][g] = p;
        rs += p;
      }
#pragma unroll
      for (int off = 1; off < 16; off <<= 1) rs += __shfl_xor(rs, off, 16);
      lrow[g] = lrow[g] * alpha + rs;
      mrow[g] = mnew;
#pragma unroll
      for (int sub = 0; sub < 4; ++sub) o[sub][g] *= alpha;
    }

    // ---- P (C-layout f32) -> per-wave LDS bf16 -> A-layout fragments ----
    {
      unsigned short* pb = &Plds[wid * 1024];
#pragma unroll
      for (int sub = 0; sub < 4; ++sub)
#pragma unroll
        for (int g = 0; g < 8; ++g)
          pb[(g + 8 * hi) * 64 + sub * 16 + lo] = f2bf(s[sub][g]);
    }
    asm volatile("s_wait_dscnt 0" ::: "memory");  // wave-local LDS RAW fence
    FragU pf[2];
#pragma unroll
    for (int kc = 0; kc < 2; ++kc) {
      pf[kc].u4[0] = *(const uint4*)&Plds[wid * 1024 + lo * 64 + kc * 32 + 8 * hi];
      pf[kc].u4[1] = *(const uint4*)&Plds[wid * 1024 + lo * 64 + kc * 32 + 16 + 8 * hi];
    }

    // ---- O += P V ----
#pragma unroll
    for (int sub = 0; sub < 4; ++sub) {
      const int n = sub * 16 + lo;  // n = d column
#pragma unroll
      for (int kc = 0; kc < 2; ++kc) {
        FragU vf;
        vf.u4[0] = *(const uint4*)&Vt[n * 80 + kc * 32 + 8 * hi];
        vf.u4[1] = *(const uint4*)&Vt[n * 80 + kc * 32 + 16 + 8 * hi];
        o[sub] = __builtin_amdgcn_wmma_f32_16x16x32_bf16(
            false, pf[kc].v, false, vf.v, (short)0, o[sub], false, false);
      }
    }
    __syncthreads();
  }

  // ---- normalize + store bf16 [B*T, 1024] ----
#pragma unroll
  for (int g = 0; g < 8; ++g) {
    const float inv = 1.f / lrow[g];
    const int t = r0 + g + 8 * hi;
    const size_t rowb = ((size_t)b * T + t) * 1024 + h * 64;
#pragma unroll
    for (int sub = 0; sub < 4; ++sub)
      Ob[rowb + sub * 16 + lo] = f2bf(o[sub][g] * inv);
  }
}

// ---------------- launch ----------------
extern "C" void kernel_launch(void* const* d_in, const int* in_sizes, int n_in,
                              void* d_out, int out_size, void* d_ws, size_t ws_size,
                              hipStream_t stream) {
  const float* x      = (const float*)d_in[0];
  const float* W_attn = (const float*)d_in[1];
  const float* b_attn = (const float*)d_in[2];
  const float* W_proj = (const float*)d_in[3];
  const float* b_proj = (const float*)d_in[4];

  const int B = 4, T = 2048, C = 1024;
  const int M = B * T;  // 8192

  char* ws = (char*)d_ws;
  unsigned short* x_bf  = (unsigned short*)ws; ws += (size_t)M * C * 2;
  unsigned short* wa_bf = (unsigned short*)ws; ws += (size_t)C * 3 * C * 2;
  unsigned short* wp_bf = (unsigned short*)ws; ws += (size_t)C * C * 2;
  unsigned short* qb    = (unsigned short*)ws; ws += (size_t)M * C * 2;
  unsigned short* kb    = (unsigned short*)ws; ws += (size_t)M * C * 2;
  unsigned short* vb    = (unsigned short*)ws; ws += (size_t)M * C * 2;
  unsigned short* attn  = (unsigned short*)ws; ws += (size_t)M * C * 2;

  cvt_f32_bf16<<<2048, 256, 0, stream>>>(x, x_bf, M * C);
  cvt_f32_bf16<<<2048, 256, 0, stream>>>(W_attn, wa_bf, C * 3 * C);
  cvt_f32_bf16<<<2048, 256, 0, stream>>>(W_proj, wp_bf, C * C);

  // qkv = x @ W_attn + b_attn  -> q,k,v [B,H,T,D] bf16
  gemm_bf16<0><<<dim3(24, 64), 256, 0, stream>>>(x_bf, wa_bf, b_attn, M, 3 * C, C,
                                                 qb, kb, vb, nullptr);
  // attention -> attn [B*T, C] bf16
  flash_attn<<<dim3(16, 64), 256, 0, stream>>>(qb, kb, vb, attn, T);
  // out = attn @ W_proj + b_proj (f32)
  gemm_bf16<1><<<dim3(8, 64), 256, 0, stream>>>(attn, wp_bf, b_proj, M, C, C,
                                                nullptr, nullptr, nullptr, (float*)d_out);
}
